// SparseMLP_7155415515700
// MI455X (gfx1250) — matrix operand order repaired and checked by reading, not compiled, not run on hardware
//
#include <hip/hip_runtime.h>
#include <stdint.h>

// ---------------- problem constants ----------------
constexpr int B   = 2;
constexpr int S   = 2048;
constexpr int D   = 1024;
constexpr int F   = 4096;
constexpr int E   = 8;
constexpr int CAP = 512;
constexpr int M_PER_E = B * CAP;      // 1024 rows per expert GEMM

// ---------------- types ----------------
typedef __attribute__((ext_vector_type(16))) __bf16 v16bf;
typedef __attribute__((ext_vector_type(8)))  float  v8f;

struct __attribute__((aligned(16))) U4 { unsigned a, b, c, d; };
struct __attribute__((aligned(8)))  U2 { unsigned a, b; };
struct __attribute__((aligned(16))) F4 { float x, y, z, w; };

union FragU { v16bf v; U4 q[2]; };

__device__ __forceinline__ unsigned short f2bf(float f) {
    union { float f; unsigned u; } c; c.f = f;
    unsigned u = c.u;
    unsigned r = u + 0x7FFFu + ((u >> 16) & 1u);   // round-to-nearest-even
    return (unsigned short)(r >> 16);
}
__device__ __forceinline__ unsigned pack2(float lo, float hi) {
    return (unsigned)f2bf(lo) | ((unsigned)f2bf(hi) << 16);
}
__device__ __forceinline__ float gelu_exact(float v) {
    return 0.5f * v * (1.0f + erff(v * 0.70710678118654752f));
}

// ---------------- K0: fp32 -> bf16 streaming convert ----------------
__global__ void cvt_bf16_kernel(const float* __restrict__ src,
                                unsigned short* __restrict__ dst, int n4) {
    int i = blockIdx.x * blockDim.x + threadIdx.x;
    int stride = gridDim.x * blockDim.x;
    for (; i < n4; i += stride) {
        F4 v = ((const F4*)src)[i];
        U2 o; o.a = pack2(v.x, v.y); o.b = pack2(v.z, v.w);
        ((U2*)dst)[i] = o;
    }
}

// ---------------- K1: router (1 wave per token) ----------------
__global__ void router_kernel(const float* __restrict__ x,
                              const float* __restrict__ wr,
                              float* __restrict__ logits_out,   // [B*S][E]
                              int*   __restrict__ eidx_ws,      // [B*S]
                              float* __restrict__ maxp_ws) {    // [B*S]
    int lane = threadIdx.x & 31;
    int wave = threadIdx.x >> 5;
    int t = blockIdx.x * 8 + wave;
    if (t >= B * S) return;

    float acc[E];
#pragma unroll
    for (int e = 0; e < E; ++e) acc[e] = 0.0f;

    const float* xp = x + (size_t)t * D;
    for (int d = lane; d < D; d += 32) {
        float xv = xp[d];
#pragma unroll
        for (int e = 0; e < E; ++e) acc[e] += xv * wr[e * D + d];
    }
#pragma unroll
    for (int e = 0; e < E; ++e)
        for (int off = 16; off >= 1; off >>= 1)
            acc[e] += __shfl_xor(acc[e], off, 32);

    if (lane == 0) {
        float lmax = acc[0]; int eidx = 0;
#pragma unroll
        for (int e = 1; e < E; ++e)
            if (acc[e] > lmax) { lmax = acc[e]; eidx = e; }
        float z = 0.0f;
#pragma unroll
        for (int e = 0; e < E; ++e) z += expf(acc[e] - lmax);
#pragma unroll
        for (int e = 0; e < E; ++e) logits_out[(size_t)t * E + e] = acc[e];
        eidx_ws[t] = eidx;
        maxp_ws[t] = 1.0f / z;   // max softmax prob
    }
}

// ---------------- K2: capacity scan (one thread per (b,e)) ----------------
__global__ void slot_kernel(const int* __restrict__ eidx_ws,
                            int*  __restrict__ tfs,          // [B][E][CAP] token-for-slot
                            int*  __restrict__ slot_kept,    // [B*S] slot or -1
                            float* __restrict__ eidx_out) {  // expert_index output (float)
    int t = threadIdx.x;
    if (t >= B * E) return;
    int b = t / E, e = t % E;
    int* tfs_be = tfs + ((size_t)b * E + e) * CAP;
    for (int c = 0; c < CAP; ++c) tfs_be[c] = -1;
    int cnt = 0;
    for (int s = 0; s < S; ++s) {
        int tok = b * S + s;
        if (eidx_ws[tok] == e) {
            cnt++;
            if (cnt <= CAP) {
                slot_kept[tok] = cnt - 1;
                tfs_be[cnt - 1] = s;
                eidx_out[tok] = (float)e;
            } else {
                slot_kept[tok] = -1;
                eidx_out[tok] = 0.0f;   // dropped -> 0
            }
        }
    }
}

// ---------------- K3: dispatch gather -> bf16 [E][B*CAP][D] ----------------
__global__ void dispatch_kernel(const float* __restrict__ x,
                                const int* __restrict__ tfs,
                                unsigned short* __restrict__ disp) {
    int r = blockIdx.x;                 // 0 .. E*B*CAP-1
    int e = r >> 10;                    // B*CAP == 1024
    int m = r & 1023;
    int b = m >> 9;                     // CAP == 512
    int c = m & 511;
    int s = tfs[((size_t)b * E + e) * CAP + c];
    unsigned short* dst = disp + ((size_t)e * M_PER_E + m) * D;
    int d0 = threadIdx.x * 4;
    U2 o;
    if (s >= 0) {
        F4 v = *(const F4*)(x + ((size_t)b * S + s) * D + d0);
        o.a = pack2(v.x, v.y); o.b = pack2(v.z, v.w);
    } else {
        o.a = 0u; o.b = 0u;
    }
    *(U2*)(dst + d0) = o;
}

// ---------------- K5: passthrough init: out = max_prob * x ----------------
__global__ void init_out_kernel(const float* __restrict__ x,
                                const float* __restrict__ maxp,
                                float* __restrict__ out) {
    int i = blockIdx.x * blockDim.x + threadIdx.x;
    int n4 = B * S * D / 4;
    int stride = gridDim.x * blockDim.x;
    for (; i < n4; i += stride) {
        int tok = i >> 8;               // D/4 == 256
        float sc = maxp[tok];
        F4 v = ((const F4*)x)[i];
        F4 o; o.x = sc * v.x; o.y = sc * v.y; o.z = sc * v.z; o.w = sc * v.w;
        ((F4*)out)[i] = o;
    }
}

// ---------------- WMMA grouped GEMM ----------------
// C[e] = A[e] (M x K, bf16, K-contiguous) * B[e]^T (N x K, bf16, K-contiguous)
// Block tile 128x128; 8 waves in a 4x2 grid, each wave owns a 32x64 sub-tile
// = 2 A-frags x 4 B-frags -> 8 accumulators of v_wmma_f32_16x16x32_bf16.
// Ping-pong LDS buffers: tile k is read while tile k+1 is committed to the
// alternate buffer and tile k+2 streams from HBM into registers.
// One __syncthreads per k-step.
template <int KDIM, int NDIM, bool FUSE_GELU>
__global__ __launch_bounds__(256)
void wmma_gemm_kernel(const unsigned short* __restrict__ A,
                      const unsigned short* __restrict__ Bm,
                      unsigned short* __restrict__ Hout,
                      float* __restrict__ Out,
                      const int* __restrict__ tfs,
                      const float* __restrict__ maxp) {
    // 128 rows x 32 bf16 (+8 pad) = 5 U4 chunks/row, 2 ping-pong buffers
    __shared__ U4 sA[2][128 * 5];
    __shared__ U4 sB[2][128 * 5];

    const int e  = blockIdx.z;
    const int m0 = blockIdx.y * 128;
    const int n0 = blockIdx.x * 128;
    const int tid  = threadIdx.x;
    const int wave = tid >> 5;
    const int lane = tid & 31;
    const int half = lane >> 4;
    const int lr   = lane & 15;
    const int wr   = wave >> 1;      // 0..3 : row group (32 rows each)
    const int wc   = wave & 1;       // 0..1 : col group (64 cols each)

    const unsigned short* Ae = A  + (size_t)e * M_PER_E * KDIM;
    const unsigned short* Be = Bm + (size_t)e * NDIM   * KDIM;

    // ---- staging (global -> regs -> LDS), 2 chunks per thread per matrix ----
    const int srow = tid >> 2;       // 0..63
    const int sc   = tid & 3;        // 16B chunk within 64B row segment
    const unsigned short* gA0 = Ae + (size_t)(m0 + srow) * KDIM + sc * 8;
    const unsigned short* gA1 = gA0 + (size_t)64 * KDIM;
    const unsigned short* gB0 = Be + (size_t)(n0 + srow) * KDIM + sc * 8;
    const unsigned short* gB1 = gB0 + (size_t)64 * KDIM;
    const int lds0 = srow * 5 + sc;
    const int lds1 = (srow + 64) * 5 + sc;

    U4 stA0, stA1, stB0, stB1;

    // ---- fragment LDS indices (constant across k) ----
    // A (16x32 bf16): lanes 0-15: K[0..7],K[16..23] -> chunks 0,2
    //                 lanes16-31: K[8..15],K[24..31] -> chunks 1,3
    const int ac0 = half ? 1 : 0;
    const int ac1 = half ? 3 : 2;
    // B (32x16): lanes 0-15 hold K[0..7],K[16..23] (chunks 0,2); lanes 16-31 K[8..15],K[24..31] (1,3)
    const int bc0 = half ? 1 : 0;
    const int bc1 = half ? 3 : 2;
    const int rowA0 = (wr * 32 + lr) * 5;
    const int rowA1 = rowA0 + 16 * 5;
    const int rowB0 = (wc * 64 + lr) * 5;

    v8f acc[8] = {};

    // prologue: tile 0 -> LDS buf0; tile 1 in flight to registers
    stA0 = *(const U4*)(gA0); stA1 = *(const U4*)(gA1);
    stB0 = *(const U4*)(gB0); stB1 = *(const U4*)(gB1);
    sA[0][lds0] = stA0; sA[0][lds1] = stA1;
    sB[0][lds0] = stB0; sB[0][lds1] = stB1;
    stA0 = *(const U4*)(gA0 + 32); stA1 = *(const U4*)(gA1 + 32);
    stB0 = *(const U4*)(gB0 + 32); stB1 = *(const U4*)(gB1 + 32);
    __syncthreads();

    int p = 0;
    for (int k0 = 0; k0 < KDIM; k0 += 32) {
        const U4* cA = sA[p];
        const U4* cB = sB[p];

        // load all fragments of tile k0, then 8 WMMA back-to-back
        FragU a0, a1, bf0, bf1, bf2, bf3;
        a0.q[0] = cA[rowA0 + ac0]; a0.q[1] = cA[rowA0 + ac1];
        a1.q[0] = cA[rowA1 + ac0]; a1.q[1] = cA[rowA1 + ac1];
        bf0.q[0] = cB[rowB0 +   0 + bc0]; bf0.q[1] = cB[rowB0 +   0 + bc1];
        bf1.q[0] = cB[rowB0 +  80 + bc0]; bf1.q[1] = cB[rowB0 +  80 + bc1];
        bf2.q[0] = cB[rowB0 + 160 + bc0]; bf2.q[1] = cB[rowB0 + 160 + bc1];
        bf3.q[0] = cB[rowB0 + 240 + bc0]; bf3.q[1] = cB[rowB0 + 240 + bc1];

        acc[0] = __builtin_amdgcn_wmma_f32_16x16x32_bf16(false, a0.v, false, bf0.v, (short)0, acc[0], false, false);
        acc[1] = __builtin_amdgcn_wmma_f32_16x16x32_bf16(false, a0.v, false, bf1.v, (short)0, acc[1], false, false);
        acc[2] = __builtin_amdgcn_wmma_f32_16x16x32_bf16(false, a0.v, false, bf2.v, (short)0, acc[2], false, false);
        acc[3] = __builtin_amdgcn_wmma_f32_16x16x32_bf16(false, a0.v, false, bf3.v, (short)0, acc[3], false, false);
        acc[4] = __builtin_amdgcn_wmma_f32_16x16x32_bf16(false, a1.v, false, bf0.v, (short)0, acc[4], false, false);
        acc[5] = __builtin_amdgcn_wmma_f32_16x16x32_bf16(false, a1.v, false, bf1.v, (short)0, acc[5], false, false);
        acc[6] = __builtin_amdgcn_wmma_f32_16x16x32_bf16(false, a1.v, false, bf2.v, (short)0, acc[6], false, false);
        acc[7] = __builtin_amdgcn_wmma_f32_16x16x32_bf16(false, a1.v, false, bf3.v, (short)0, acc[7], false, false);

        // commit tile k0+32 (in registers) to the inactive buffer; its previous
        // readers all retired at the barrier that ended iteration k0-32.
        if (k0 + 32 < KDIM) {
            U4* nA = sA[p ^ 1];
            U4* nB = sB[p ^ 1];
            nA[lds0] = stA0; nA[lds1] = stA1;
            nB[lds0] = stB0; nB[lds1] = stB1;
            if (k0 + 64 < KDIM) {   // stage tile k0+64 from HBM
                const int kn = k0 + 64;
                stA0 = *(const U4*)(gA0 + kn); stA1 = *(const U4*)(gA1 + kn);
                stB0 = *(const U4*)(gB0 + kn); stB1 = *(const U4*)(gB1 + kn);
            }
        }
        __syncthreads();
        p ^= 1;
    }

    // Epilogue. C/D layout: lane -> n = lr; VGPR r -> m = r + 8*half.
    if constexpr (FUSE_GELU) {
        unsigned short* He = Hout + (size_t)e * M_PER_E * NDIM;
#pragma unroll
        for (int ar = 0; ar < 2; ++ar) {
#pragma unroll
            for (int r = 0; r < 8; ++r) {
                int m = m0 + wr * 32 + ar * 16 + half * 8 + r;
#pragma unroll
                for (int bj = 0; bj < 4; ++bj) {
                    int n = n0 + wc * 64 + bj * 16 + lr;
                    He[(size_t)m * NDIM + n] = f2bf(gelu_exact(acc[ar * 4 + bj][r]));
                }
            }
        }
    } else {
#pragma unroll
        for (int ar = 0; ar < 2; ++ar) {
#pragma unroll
            for (int r = 0; r < 8; ++r) {
                int m = m0 + wr * 32 + ar * 16 + half * 8 + r;
                int b = m >> 9;          // CAP == 512
                int c = m & 511;
                int s = tfs[((size_t)b * E + e) * CAP + c];
                if (s < 0) continue;     // empty slot: output already holds passthrough
                int tok = b * S + s;
                float scale = maxp[tok];
                float* orow = Out + (size_t)tok * D;
#pragma unroll
                for (int bj = 0; bj < 4; ++bj) {
                    int n = n0 + wc * 64 + bj * 16 + lr;
                    orow[n] = scale * acc[ar * 4 + bj][r];
                }
            }
        }
    }
}

// ---------------- host launcher ----------------
extern "C" void kernel_launch(void* const* d_in, const int* in_sizes, int n_in,
                              void* d_out, int out_size, void* d_ws, size_t ws_size,
                              hipStream_t stream) {
    const float* x  = (const float*)d_in[0];   // [B,S,D]
    const float* wr = (const float*)d_in[1];   // [E,D]
    const float* wi = (const float*)d_in[2];   // [E,F,D]
    const float* wo = (const float*)d_in[3];   // [E,D,F]

    float* out      = (float*)d_out;                                   // [B,S,D]
    float* logits   = out + (size_t)B * S * D;                         // [B,S,E]
    float* eidx_out = logits + (size_t)B * S * E;                      // [B,S]

    // workspace layout
    char* ws = (char*)d_ws;
    unsigned short* wi_bf   = (unsigned short*)ws;                     ws += (size_t)E * F * D * 2;
    unsigned short* wo_bf   = (unsigned short*)ws;                     ws += (size_t)E * D * F * 2;
    unsigned short* disp_bf = (unsigned short*)ws;                     ws += (size_t)E * M_PER_E * D * 2;
    unsigned short* h_bf    = (unsigned short*)ws;                     ws += (size_t)E * M_PER_E * F * 2;
    int*   eidx_ws  = (int*)ws;                                        ws += (size_t)B * S * 4;
    int*   slot_ws  = (int*)ws;                                        ws += (size_t)B * S * 4;
    float* maxp_ws  = (float*)ws;                                      ws += (size_t)B * S * 4;
    int*   tfs_ws   = (int*)ws;                                        ws += (size_t)B * E * CAP * 4;

    // K0: weight conversion (2 x 33.5M elems, 4 per thread)
    cvt_bf16_kernel<<<2048, 256, 0, stream>>>(wi, wi_bf, E * F * D / 4);
    cvt_bf16_kernel<<<2048, 256, 0, stream>>>(wo, wo_bf, E * D * F / 4);

    // K1: router — 8 tokens (waves) per 256-thread block
    router_kernel<<<(B * S) / 8, 256, 0, stream>>>(x, wr, logits, eidx_ws, maxp_ws);

    // K2: sequential capacity scan, 16 active threads
    slot_kernel<<<1, 32, 0, stream>>>(eidx_ws, tfs_ws, slot_ws, eidx_out);

    // K3: dispatch gather -> bf16 capacity buffers
    dispatch_kernel<<<E * M_PER_E, 256, 0, stream>>>(x, tfs_ws, disp_bf);

    // K4: h = gelu(disp @ wi^T)   M=1024, K=1024, N=4096 per expert
    wmma_gemm_kernel<D, F, true><<<dim3(F / 128, M_PER_E / 128, E), 256, 0, stream>>>(
        disp_bf, wi_bf, h_bf, nullptr, nullptr, nullptr);

    // K5: out = max_prob * x (passthrough for dropped tokens)
    init_out_kernel<<<2048, 256, 0, stream>>>(x, maxp_ws, out);

    // K6: eout = h @ wo^T, scatter kept tokens scaled by max_prob
    wmma_gemm_kernel<F, D, false><<<dim3(D / 128, M_PER_E / 128, E), 256, 0, stream>>>(
        h_bf, wo_bf, nullptr, out, tfs_ws, maxp_ws);
}
